// SemGCN_Heatmaps_28003186770006
// MI455X (gfx1250) — compile-verified, matrix operand order checked by reading, not run on hardware
//
#include <hip/hip_runtime.h>
#include <hip/hip_bf16.h>
#include <math.h>

// ---------------------------------------------------------------------------
// SemGCN + heatmap joint feature extraction for MI455X (gfx1250).
// Matmuls use V_WMMA_F32_16X16X32_F16 (fp32 accumulate, f16 operands).
// ---------------------------------------------------------------------------

typedef __attribute__((ext_vector_type(16))) _Float16 v16h;
typedef __attribute__((ext_vector_type(8)))  float    v8f;
typedef __attribute__((ext_vector_type(4)))  float    f4;

#define NJ 17
#define BB 64
#define MROWS (BB * NJ)      // 1088 rows, = 68 tiles of 16
#define STRIDE 640           // activation row stride (max feature width)
#define BN_EPS 1e-5f

__constant__ int c_edges[16][2] = {
    {0,1},{1,2},{2,3},{0,4},{4,5},{5,6},{0,7},{7,8},{8,9},{9,10},
    {8,11},{11,12},{12,13},{8,14},{14,15},{15,16}};

// ---------------------------------------------------------------------------
// 1) Joint feature extraction:  jf[b,j,c] = (1/HW)*sum_hw f*hm  +  max_hw f*hm
//    One block per (b,c); threads stride over hw (coalesced feature reads).
// ---------------------------------------------------------------------------
__global__ void extract_feats_kernel(const float* __restrict__ feat,
                                     const float* __restrict__ hm,
                                     float* __restrict__ jf,
                                     int C, int HW, int wshift, int step) {
    const int tid = threadIdx.x;
    const int b = blockIdx.x / C;
    const int c = blockIdx.x % C;
    const int wmask = (1 << wshift) - 1;
    const float* f = feat + ((size_t)b * C + c) * HW;
    const float* h = hm + (size_t)b * NJ * 4096;

    float sum[NJ];
    float mx[NJ];
#pragma unroll
    for (int j = 0; j < NJ; j++) { sum[j] = 0.f; mx[j] = -3.0e38f; }

    for (int p = tid; p < HW; p += 256) {
        const int r = p >> wshift;
        const int col = p & wmask;
        const int hidx = (r * step) * 64 + col * step;   // nearest resize
        const float fv = f[p];
#pragma unroll
        for (int j = 0; j < NJ; j++) {
            const float t = fv * h[j * 4096 + hidx];
            sum[j] += t;
            mx[j] = fmaxf(mx[j], t);
        }
    }

    __shared__ float red[256];
    const float inv = 1.0f / (float)HW;
    for (int j = 0; j < NJ; j++) {
        red[tid] = sum[j];
        __syncthreads();
        for (int o = 128; o > 0; o >>= 1) {
            if (tid < o) red[tid] += red[tid + o];
            __syncthreads();
        }
        const float stot = red[0];
        __syncthreads();
        red[tid] = mx[j];
        __syncthreads();
        for (int o = 128; o > 0; o >>= 1) {
            if (tid < o) red[tid] = fmaxf(red[tid], red[tid + o]);
            __syncthreads();
        }
        if (tid == 0)
            jf[((size_t)b * NJ + j) * C + c] = stot * inv + red[0];
        __syncthreads();
    }
}

// ---------------------------------------------------------------------------
// 2a) Fast WMMA GEMM: requires K%32==0, N%32==0, lda/ldw/ldc row strides.
//     One wave -> 16x32 output (two 16x16 accumulators sharing the A frag).
//     A loads are unconditional float4 vector loads (rows contiguous in K).
// ---------------------------------------------------------------------------
__global__ void gemm_wmma_fast(const float* __restrict__ A, int lda,
                               const float* __restrict__ Wm, int ldw,
                               float* __restrict__ C, int ldc, int K) {
    const int lane = threadIdx.x & 31;
    const int half = lane >> 4;                 // 0 or 1
    const int mrow = blockIdx.x * 16 + (lane & 15);
    const int ncol0 = blockIdx.y * 32 + (lane & 15);
    const int ncol1 = ncol0 + 16;

    const float* arow = A + (size_t)mrow * lda;

    v8f acc0 = {0.f, 0.f, 0.f, 0.f, 0.f, 0.f, 0.f, 0.f};
    v8f acc1 = {0.f, 0.f, 0.f, 0.f, 0.f, 0.f, 0.f, 0.f};

    for (int k0 = 0; k0 < K; k0 += 32) {
        const int kb = k0 + half * 8;           // lane's first K
        // A fragment: halves e=0..7 <- k=kb..kb+7, e=8..15 <- k=kb+16..kb+23
        const f4 a0 = *(const f4*)(arow + kb);
        const f4 a1 = *(const f4*)(arow + kb + 4);
        const f4 a2 = *(const f4*)(arow + kb + 16);
        const f4 a3 = *(const f4*)(arow + kb + 20);
        v16h a;
#pragma unroll
        for (int e = 0; e < 4; e++) {
            a[e]      = (_Float16)a0[e];
            a[e + 4]  = (_Float16)a1[e];
            a[e + 8]  = (_Float16)a2[e];
            a[e + 12] = (_Float16)a3[e];
        }
        v16h b0, b1;
#pragma unroll
        for (int e = 0; e < 16; e++) {
            const int kk = kb + (e < 8 ? e : e + 8);
            const float* wrow = Wm + (size_t)kk * ldw;
            b0[e] = (_Float16)wrow[ncol0];
            b1[e] = (_Float16)wrow[ncol1];
        }
        acc0 = __builtin_amdgcn_wmma_f32_16x16x32_f16(
            false, a, false, b0, (short)0, acc0, false, false);
        acc1 = __builtin_amdgcn_wmma_f32_16x16x32_f16(
            false, a, false, b1, (short)0, acc1, false, false);
    }

#pragma unroll
    for (int r = 0; r < 8; r++) {
        const int m = blockIdx.x * 16 + r + half * 8;
        C[(size_t)m * ldc + ncol0] = acc0[r];
        C[(size_t)m * ldc + ncol1] = acc1[r];
    }
}

// ---------------------------------------------------------------------------
// 2b) Guarded WMMA GEMM (input layer K=2; output heads N=3/2).
//     Addresses are clamped so loads are unconditional; OOB values selected
//     to zero with cndmask (no exec-mask branches).
// ---------------------------------------------------------------------------
__global__ void gemm_wmma_guarded(const float* __restrict__ A, int lda,
                                  const float* __restrict__ Wm, int ldw,
                                  float* __restrict__ C, int ldc,
                                  int N, int K) {
    const int lane = threadIdx.x & 31;
    const int half = lane >> 4;
    const int mrow = blockIdx.x * 16 + (lane & 15);
    const int ncol = blockIdx.y * 16 + (lane & 15);
    const int ncolc = ncol < N - 1 ? ncol : N - 1;
    const bool colok = ncol < N;

    const float* arow = A + (size_t)mrow * lda;

    v8f acc = {0.f, 0.f, 0.f, 0.f, 0.f, 0.f, 0.f, 0.f};

    for (int k0 = 0; k0 < K; k0 += 32) {
        const int kb = k0 + half * 8;
        v16h a, bf;
#pragma unroll
        for (int e = 0; e < 16; e++) {
            const int kk = kb + (e < 8 ? e : e + 8);
            const int kkc = kk < K - 1 ? kk : K - 1;
            const bool kok = kk < K;
            float av = arow[kkc];
            float bv = Wm[(size_t)kkc * ldw + ncolc];
            av = kok ? av : 0.f;
            bv = (kok && colok) ? bv : 0.f;
            a[e]  = (_Float16)av;
            bf[e] = (_Float16)bv;
        }
        acc = __builtin_amdgcn_wmma_f32_16x16x32_f16(
            false, a, false, bf, (short)0, acc, false, false);
    }

#pragma unroll
    for (int r = 0; r < 8; r++) {
        const int m = blockIdx.x * 16 + r + half * 8;
        if (colok) C[(size_t)m * ldc + ncol] = acc[r];
    }
}

// ---------------------------------------------------------------------------
// 3) Masked row-softmax over learned edge logits -> adjacency [17,17]
// ---------------------------------------------------------------------------
__global__ void adj_kernel(const float* __restrict__ e, float* __restrict__ adj) {
    const int i = threadIdx.x;
    if (i >= NJ) return;
    unsigned m = 1u << i;
    for (int t = 0; t < 16; t++) {
        if (c_edges[t][0] == i) m |= 1u << c_edges[t][1];
        if (c_edges[t][1] == i) m |= 1u << c_edges[t][0];
    }
    float mx = -3.0e38f;
    for (int j = 0; j < NJ; j++)
        if ((m >> j) & 1) mx = fmaxf(mx, e[i * NJ + j]);
    float s = 0.f;
    for (int j = 0; j < NJ; j++)
        if ((m >> j) & 1) s += __expf(e[i * NJ + j] - mx);
    const float inv = 1.0f / s;
    for (int j = 0; j < NJ; j++)
        adj[i * NJ + j] = ((m >> j) & 1) ? __expf(e[i * NJ + j] - mx) * inv : 0.f;
}

// ---------------------------------------------------------------------------
// 4) Gconv epilogue: out[b,i,:] = adj[i,i]*h0[b,i,:] + sum_{j!=i} adj[i,j]*h1[b,j,:]
//                    + bias  (+ BN scale/shift + ReLU when use_bn)
// ---------------------------------------------------------------------------
__global__ void mix_kernel(const float* __restrict__ h0,
                           const float* __restrict__ h1, int ldh,
                           const float* __restrict__ adj,
                           const float* __restrict__ bias,
                           const float* __restrict__ gamma,
                           const float* __restrict__ beta, int use_bn,
                           float* __restrict__ out, int ldo, int D) {
    const int row = blockIdx.x;            // b*17 + i
    const int i = row % NJ;
    const int d = blockIdx.y * 256 + threadIdx.x;
    __shared__ float arow[NJ];
    if (threadIdx.x < NJ) arow[threadIdx.x] = adj[i * NJ + threadIdx.x];
    __syncthreads();
    if (d >= D) return;

    const int b = row / NJ;
    float v = arow[i] * h0[((size_t)b * NJ + i) * ldh + d];
#pragma unroll
    for (int j = 0; j < NJ; j++) {
        if (j == i) continue;
        v += arow[j] * h1[((size_t)b * NJ + j) * ldh + d];
    }
    v += bias[d];
    if (use_bn) {
        v = v * (gamma[d] * rsqrtf(1.0f + BN_EPS)) + beta[d];
        v = fmaxf(v, 0.f);
    }
    out[(size_t)row * ldo + d] = v;
}

// ---------------------------------------------------------------------------
// 5) Node self-attention core: scores (17 dots), softmax, a @ v.
//    One block per (b,i).
// ---------------------------------------------------------------------------
__global__ void attn_kernel(const float* __restrict__ q,
                            const float* __restrict__ k,
                            const float* __restrict__ v, int ld,
                            float* __restrict__ out, int ldo, int D) {
    const int tid = threadIdx.x;
    const int row = blockIdx.x;     // b*17 + i
    const int b = row / NJ;

    float part[NJ];
#pragma unroll
    for (int j = 0; j < NJ; j++) part[j] = 0.f;

    for (int d = tid; d < D; d += 256) {
        const float qv = q[(size_t)row * ld + d];
#pragma unroll
        for (int j = 0; j < NJ; j++)
            part[j] += qv * k[((size_t)b * NJ + j) * ld + d];
    }

    __shared__ float red[256];
    __shared__ float sc[NJ];
    __shared__ float aw[NJ];
    for (int j = 0; j < NJ; j++) {
        red[tid] = part[j];
        __syncthreads();
        for (int o = 128; o > 0; o >>= 1) {
            if (tid < o) red[tid] += red[tid + o];
            __syncthreads();
        }
        if (tid == 0) sc[j] = red[0];
        __syncthreads();
    }
    if (tid == 0) {
        const float scale = rsqrtf((float)D);
        float mx = -3.0e38f;
        for (int j = 0; j < NJ; j++) mx = fmaxf(mx, sc[j] * scale);
        float s = 0.f;
        for (int j = 0; j < NJ; j++) {
            const float ex = __expf(sc[j] * scale - mx);
            aw[j] = ex;
            s += ex;
        }
        const float inv = 1.0f / s;
        for (int j = 0; j < NJ; j++) aw[j] *= inv;
    }
    __syncthreads();

    for (int d = tid; d < D; d += 256) {
        float acc = 0.f;
#pragma unroll
        for (int j = 0; j < NJ; j++)
            acc += aw[j] * v[((size_t)b * NJ + j) * ld + d];
        out[(size_t)row * ldo + d] = acc;
    }
}

// ---------------------------------------------------------------------------
// 6) Residual fuse: x += g2 + wo   (attention internal residual + block residual)
// ---------------------------------------------------------------------------
__global__ void add3_kernel(float* __restrict__ x,
                            const float* __restrict__ a,
                            const float* __restrict__ b, int ld, int D) {
    const int row = blockIdx.x;
    const int d = blockIdx.y * 256 + threadIdx.x;
    if (d < D) {
        const size_t idx = (size_t)row * ld + d;
        x[idx] = x[idx] + a[idx] + b[idx];
    }
}

// 7) Concat [jf | x] along feature dim
__global__ void concat_kernel(const float* __restrict__ jf, int Cj,
                              const float* __restrict__ x, int ldx, int Dx,
                              float* __restrict__ out, int ldo) {
    const int row = blockIdx.x;
    const int d = blockIdx.y * 256 + threadIdx.x;
    if (d < Cj + Dx) {
        const float v = (d < Cj) ? jf[(size_t)row * Cj + d]
                                 : x[(size_t)row * ldx + (d - Cj)];
        out[(size_t)row * ldo + d] = v;
    }
}

// ---------------------------------------------------------------------------
// Host orchestration
// ---------------------------------------------------------------------------
// flattened input indices (setup_inputs insertion order, recursive):
#define IN_X    0
#define IN_HM   1
#define IN_F0   2
#define IN_F1   3
#define IN_F2   4
#define P_INPUT 5    // gblock: W,e,b,gamma,beta
#define P_RES1  10   // res: g1(5), g2(5), Wq,Wk,Wv,Wo
#define P_RES2  24
#define P_RES3  38
#define P_RES4  52
#define P_OUT1  66   // sem: W,e,b
#define P_OUT2  69
#define P_OUT3  72

extern "C" void kernel_launch(void* const* d_in, const int* in_sizes, int n_in,
                              void* d_out, int out_size, void* d_ws, size_t ws_size,
                              hipStream_t stream) {
    (void)in_sizes; (void)n_in; (void)out_size; (void)ws_size;

    auto F = [&](int i) { return (const float*)d_in[i]; };

    float* ws = (float*)d_ws;
    const size_t MD = (size_t)MROWS * STRIDE;
    float* jf0  = ws;                       // 64*17*256
    float* jf1  = jf0 + (size_t)BB * NJ * 256;
    float* jf2  = jf1 + (size_t)BB * NJ * 128;
    float* adjw = jf2 + (size_t)BB * NJ * 128;   // 289 (+pad)
    float* ACT0 = adjw + 512;
    float* ACT1 = ACT0 + MD;
    float* G1   = ACT1 + MD;
    float* G2   = G1 + MD;
    float* Qb   = G2 + MD;
    float* Kb   = Qb + MD;
    float* Vb   = Kb + MD;
    float* AT   = Vb + MD;
    float* WO   = AT + MD;

    // -------- phase A: joint features from the three pyramids ------------
    extract_feats_kernel<<<dim3(BB * 256), 256, 0, stream>>>(
        F(IN_F0), F(IN_HM), jf0, 256, 4096, 6, 1);
    extract_feats_kernel<<<dim3(BB * 128), 256, 0, stream>>>(
        F(IN_F1), F(IN_HM), jf1, 128, 1024, 5, 2);
    extract_feats_kernel<<<dim3(BB * 128), 256, 0, stream>>>(
        F(IN_F2), F(IN_HM), jf2, 128, 256, 4, 4);

    auto gemm = [&](const float* A, int lda, const float* Wm, int ldw,
                    float* C, int ldc, int N, int K) {
        if ((N % 32 == 0) && (K % 32 == 0)) {
            dim3 g(MROWS / 16, N / 32);
            gemm_wmma_fast<<<g, 32, 0, stream>>>(A, lda, Wm, ldw, C, ldc, K);
        } else {
            dim3 g(MROWS / 16, (N + 15) / 16);
            gemm_wmma_guarded<<<g, 32, 0, stream>>>(A, lda, Wm, ldw, C, ldc, N, K);
        }
    };

    // sem graph conv (+ optional BN/ReLU).  Uses Qb/Kb as h0/h1 scratch.
    auto gconv = [&](const float* xp, int lda, int din, int dout, int pbase,
                     float* outp, int ldo, bool bn) {
        adj_kernel<<<1, 32, 0, stream>>>(F(pbase + 1), adjw);
        const float* Wp = F(pbase + 0);
        gemm(xp, lda, Wp, dout, Qb, STRIDE, dout, din);                         // h0
        gemm(xp, lda, Wp + (size_t)din * dout, dout, Kb, STRIDE, dout, din);    // h1
        dim3 g(MROWS, (dout + 255) / 256);
        mix_kernel<<<g, 256, 0, stream>>>(Qb, Kb, STRIDE, adjw, F(pbase + 2),
                                          bn ? F(pbase + 3) : (const float*)nullptr,
                                          bn ? F(pbase + 4) : (const float*)nullptr,
                                          bn ? 1 : 0, outp, ldo, dout);
    };

    // residual block (din == dout after concat for all res blocks)
    auto resblock = [&](float* xact, int D, int pbase) {
        gconv(xact, STRIDE, D, D, pbase + 0, G1, STRIDE, true);
        gconv(G1, STRIDE, D, D, pbase + 5, G2, STRIDE, true);
        gemm(G2, STRIDE, F(pbase + 10), D, Qb, STRIDE, D, D);   // q
        gemm(G2, STRIDE, F(pbase + 11), D, Kb, STRIDE, D, D);   // k
        gemm(G2, STRIDE, F(pbase + 12), D, Vb, STRIDE, D, D);   // v
        attn_kernel<<<dim3(MROWS), 256, 0, stream>>>(Qb, Kb, Vb, STRIDE,
                                                     AT, STRIDE, D);
        gemm(AT, STRIDE, F(pbase + 13), D, WO, STRIDE, D, D);   // (a@v)@Wo
        dim3 g(MROWS, (D + 255) / 256);
        add3_kernel<<<g, 256, 0, stream>>>(xact, G2, WO, STRIDE, D);
    };

    auto concat = [&](const float* jf, int Cj, const float* x, int Dx, float* out) {
        dim3 g(MROWS, (Cj + Dx + 255) / 256);
        concat_kernel<<<g, 256, 0, stream>>>(jf, Cj, x, STRIDE, Dx, out, STRIDE);
    };

    float* outp = (float*)d_out;

    // -------- phase B: the graph network ----------------------------------
    gconv(F(IN_X), 2, 2, 128, P_INPUT, ACT0, STRIDE, true);     // input block
    resblock(ACT0, 128, P_RES1);

    concat(jf0, 256, ACT0, 128, ACT1);                          // -> 384
    resblock(ACT1, 384, P_RES2);
    gconv(ACT1, STRIDE, 384, 3, P_OUT1, outp, 3, false);        // out1 [64,17,3]

    concat(jf1, 128, ACT1, 384, ACT0);                          // -> 512
    resblock(ACT0, 512, P_RES3);
    gconv(ACT0, STRIDE, 512, 3, P_OUT2, outp + 3264, 3, false); // out2 [64,17,3]

    concat(jf2, 128, ACT0, 512, ACT1);                          // -> 640
    resblock(ACT1, 640, P_RES4);
    gconv(ACT1, STRIDE, 640, 2, P_OUT3, outp + 6528, 2, false); // out3 [64,17,2]
}
